// Decoder_68573447848532
// MI455X (gfx1250) — matrix operand order repaired and checked by reading, not compiled, not run on hardware
//
#include <hip/hip_runtime.h>
#include <hip/hip_bf16.h>
#include <cstdint>

// ---------------- model dims ----------------
#define Bk   16
#define Tk   512
#define Lk   128
#define ENCk 512
#define Hk   1024
#define Ek   512
#define Vk   10000
#define Ak   512
#define STEPS 129   // L+1

typedef __attribute__((ext_vector_type(16))) __bf16 v16bf;
typedef __attribute__((ext_vector_type(8)))  __bf16 v8bf;
typedef __attribute__((ext_vector_type(8)))  float  v8f;

__device__ __forceinline__ unsigned short f2bf(float f) {
  unsigned u = __float_as_uint(f);
  u += 0x7FFFu + ((u >> 16) & 1u);     // round to nearest even
  return (unsigned short)(u >> 16);
}
__device__ __forceinline__ float bf2f(unsigned short s) {
  return __uint_as_float(((unsigned)s) << 16);
}
__device__ __forceinline__ float sigm(float x) { return 1.f / (1.f + __expf(-x)); }

// ---------------------------------------------------------------------------
// Skinny-M WMMA GEMM: out[m0+0..15][n] = sum_k X[m][k]*WT[n][k] + bias
//  - A tile [16][K] bf16 is async-staged to LDS ONCE per block
//    (global_load_async_to_lds_b128, ASYNCcnt) and shared by all 8 waves.
//  - Each wave owns one 16x16 output tile, streams K in steps of 32 via
//    v_wmma_f32_16x16x32_bf16 (B fed straight from L2, A from LDS).
// ---------------------------------------------------------------------------
__global__ void __launch_bounds__(256)
wmma_gemm16_k(const unsigned short* __restrict__ X, int lda,
              const unsigned short* __restrict__ WT,
              const float* __restrict__ bias,
              float* __restrict__ outF,
              unsigned short* __restrict__ outB,
              int ldo, int N, int K, int act)
{
  __shared__ unsigned short sA[16 * 2048];          // up to 64 KB (K<=2048)
  int m0 = blockIdx.y * 16;

  // ---- async stage A[16][K] -> LDS (16B per lane per issue) ----
  int cpr = K >> 3;                                  // 8-ushort chunks per row
  for (int r = 0; r < 16; ++r) {
    const unsigned short* src = X + (size_t)(m0 + r) * lda;
    unsigned short* dst = sA + r * K;
    for (int c = threadIdx.x; c < cpr; c += 256) {
      asm volatile("global_load_async_to_lds_b128 %0, %1, off"
                   :: "v"((unsigned)(uintptr_t)(dst + c * 8)),
                      "v"(src + c * 8)
                   : "memory");
    }
  }
  asm volatile("s_wait_asynccnt 0x0" ::: "memory");  // my async fills landed
  __syncthreads();                                   // everyone's fills landed

  int wave = threadIdx.x >> 5;
  int lane = threadIdx.x & 31;
  int n0 = (blockIdx.x * 8 + wave) * 16;
  if (n0 < N) {                                      // wave-uniform guard
    int col = lane & 15;
    int hi  = lane >> 4;
    // A k order used here (contiguous, same as B): lane<16 holds K+0..15 ; lane>=16 K+16..31
    const unsigned short* aB = sA + col * K + hi * 16;
    // B k order used here (contiguous, same as A): col = lane&15; lanes 0-15 K=0..15, 16-31 K=16..31
    const unsigned short* wB = WT + (size_t)(n0 + col) * K + hi * 16;

    v8f acc = {0.f,0.f,0.f,0.f,0.f,0.f,0.f,0.f};
    for (int k = 0; k < K; k += 32) {
      v8bf alo = *(const v8bf*)(aB + k);             // ds_load_b128
      v8bf ahi = *(const v8bf*)(aB + k + 8);
      v16bf a  = __builtin_shufflevector(alo, ahi,
                   0,1,2,3,4,5,6,7,8,9,10,11,12,13,14,15);
      v16bf b  = *(const v16bf*)(wB + k);            // 2x global_load_b128
      acc = __builtin_amdgcn_wmma_f32_16x16x32_bf16(false, a, false, b,
                                                    (short)0, acc, false, false);
    }

    int n = n0 + col;
    float bv = bias ? bias[n] : 0.f;
#pragma unroll
    for (int v = 0; v < 8; ++v) {
      int m = m0 + v + hi * 8;                       // D: VGPR v -> row (+8 hi)
      float r = acc[v] + bv;
      if (act == 1) r = tanhf(r);
      if (outF) outF[(size_t)m * ldo + n] = r;
      if (outB) outB[(size_t)m * ldo + n] = f2bf(r);
    }
  }
}

// ---------------- one-time setup kernels ----------------
__global__ void pack_lstm_k(const float* __restrict__ Wih, const float* __restrict__ Whh,
                            unsigned short* __restrict__ dst)   // [4096][2048] bf16
{
  size_t idx = (size_t)blockIdx.x * 256 + threadIdx.x;          // 4096*2048
  int k = (int)(idx & 2047); size_t n = idx >> 11;
  float v = (k < Hk) ? Wih[n * Hk + k] : Whh[n * Hk + (k - Hk)];
  dst[idx] = f2bf(v);
}

__global__ void transpose_bf_k(const float* __restrict__ src,   // [K][N]
                               unsigned short* __restrict__ dst,// [N][K]
                               int K, int N, long total)
{
  long idx = (long)blockIdx.x * 256 + threadIdx.x;
  if (idx >= total) return;
  int k = (int)(idx % K); long n = idx / K;
  dst[idx] = f2bf(src[(size_t)k * N + n]);
}

__global__ void cvt_bf_k(const float* __restrict__ src, unsigned short* __restrict__ dst,
                         long total)
{
  long idx = (long)blockIdx.x * 256 + threadIdx.x;
  if (idx >= total) return;
  dst[idx] = f2bf(src[idx]);
}

__global__ void gather_emb_k(const int* __restrict__ ys, const float* __restrict__ embed,
                             unsigned short* __restrict__ yemb) // [129][16][512]
{
  int idx = blockIdx.x * 256 + threadIdx.x;
  if (idx >= STEPS * Bk * Ek) return;
  int e  = idx & (Ek - 1);
  int bt = idx >> 9;
  int b  = bt & (Bk - 1);
  int t  = bt >> 4;
  int tok = (t == 0) ? 1 : ys[b * Lk + (t - 1)];                // SOS = 1
  yemb[idx] = f2bf(embed[(size_t)tok * Ek + e]);
}

__global__ void init_state_k(float* c0, float* h0, float* c1, float* h1,
                             float* ctx, unsigned short* x1)
{
  int i = blockIdx.x * 256 + threadIdx.x;                       // 16*1024
  c0[i] = 0.f; h0[i] = 0.f; c1[i] = 0.f; h1[i] = 0.f;
  int b = i >> 10, n = i & 1023;
  x1[(size_t)b * 2048 + 1024 + n] = 0;                          // bf16(h1=0)
  if (i < Bk * ENCk) ctx[i] = 0.f;
}

// ---------------- per-step kernels ----------------
__global__ void build_x0_k(const unsigned short* __restrict__ ye,  // [16][512] this step
                           const float* __restrict__ ctx,          // [16][512]
                           const float* __restrict__ h0,           // [16][1024]
                           unsigned short* __restrict__ x0)        // [16][2048]
{
  int idx = blockIdx.x * 256 + threadIdx.x;                        // 16*2048
  int b = idx >> 11, k = idx & 2047;
  unsigned short v;
  if (k < Ek)             v = ye[b * Ek + k];
  else if (k < Ek + ENCk) v = f2bf(ctx[b * ENCk + (k - Ek)]);
  else                    v = f2bf(h0[b * Hk + (k - Ek - ENCk)]);
  x0[idx] = v;
}

__global__ void lstm0_update_k(const float* __restrict__ g, float* __restrict__ c,
                               float* __restrict__ h, unsigned short* __restrict__ x1)
{
  int idx = blockIdx.x * 256 + threadIdx.x;                        // 16*1024
  int b = idx >> 10, n = idx & 1023;
  const float* gb = g + (size_t)b * 4 * Hk;
  float i = gb[n], f = gb[n + Hk], gg = gb[n + 2 * Hk], o = gb[n + 3 * Hk];
  float cc = sigm(f) * c[idx] + sigm(i) * tanhf(gg);
  float hh = sigm(o) * tanhf(cc);
  c[idx] = cc; h[idx] = hh;
  x1[(size_t)b * 2048 + n] = f2bf(hh);                             // x1 = [h0, h1_old]
}

__global__ void lstm1_update_k(const float* __restrict__ g, float* __restrict__ c,
                               float* __restrict__ h, const float* __restrict__ h0,
                               unsigned short* __restrict__ x1,
                               unsigned short* __restrict__ xbn)
{
  int idx = blockIdx.x * 256 + threadIdx.x;
  int b = idx >> 10, n = idx & 1023;
  const float* gb = g + (size_t)b * 4 * Hk;
  float i = gb[n], f = gb[n + Hk], gg = gb[n + 2 * Hk], o = gb[n + 3 * Hk];
  float cc = sigm(f) * c[idx] + sigm(i) * tanhf(gg);
  float hh = sigm(o) * tanhf(cc);
  c[idx] = cc; h[idx] = hh;
  x1[(size_t)b * 2048 + 1024 + n] = f2bf(hh);                      // for next step
  float dec = hh + h0[idx];                                        // residual
  xbn[(size_t)b * 1536 + n] = f2bf(dec);                           // xbn = [dec, ctx]
}

// e[b,t] = sum_a tanh(enc_proj[b,t,a]+b_att[a] + decW[b,a]) * v_att[a]
__global__ void att_e_k(const unsigned short* __restrict__ epb,
                        const float* __restrict__ decW,
                        const float* __restrict__ v_att,
                        float* __restrict__ e)
{
  int wave = blockIdx.x * 8 + (threadIdx.x >> 5);                  // 8192 waves
  int lane = threadIdx.x & 31;
  int b = wave >> 9, t = wave & (Tk - 1);
  const unsigned short* ep = epb + ((size_t)(b * Tk + t)) * Ak + lane * 16;
  const float* dw = decW + b * Ak + lane * 16;
  const float* va = v_att + lane * 16;
  float s = 0.f;
#pragma unroll
  for (int i = 0; i < 16; ++i)
    s += tanhf(bf2f(ep[i]) + dw[i]) * va[i];
  for (int off = 16; off > 0; off >>= 1) s += __shfl_down(s, off, 32);
  if (lane == 0) e[b * Tk + t] = s;
}

__global__ void att_softmax_k(const float* __restrict__ e, const int* __restrict__ lens,
                              float* __restrict__ aw)
{
  __shared__ float red[256];
  int b = blockIdx.x, tid = threadIdx.x;
  int len = lens[b];
  float v0 = e[b * Tk + tid];
  float v1 = e[b * Tk + 256 + tid];
  if (tid >= len)       v0 = -1e30f;
  if (256 + tid >= len) v1 = -1e30f;
  red[tid] = fmaxf(v0, v1); __syncthreads();
  for (int s = 128; s; s >>= 1) { if (tid < s) red[tid] = fmaxf(red[tid], red[tid + s]); __syncthreads(); }
  float m = red[0]; __syncthreads();
  float e0 = (tid < len)       ? __expf(v0 - m) : 0.f;
  float e1 = (256 + tid < len) ? __expf(v1 - m) : 0.f;
  red[tid] = e0 + e1; __syncthreads();
  for (int s = 128; s; s >>= 1) { if (tid < s) red[tid] += red[tid + s]; __syncthreads(); }
  float inv = 1.f / red[0];
  aw[b * Tk + tid]       = e0 * inv;
  aw[b * Tk + 256 + tid] = e1 * inv;
}

__global__ void att_ctx_k(const float* __restrict__ aw,
                          const unsigned short* __restrict__ eob,  // bf16 enc_out
                          float* __restrict__ ctx,
                          unsigned short* __restrict__ xbn)
{
  __shared__ float s_aw[Tk];
  int b = blockIdx.x, e = threadIdx.x;                             // 512 threads
  s_aw[e] = aw[b * Tk + e];
  __syncthreads();
  const unsigned short* eo = eob + (size_t)b * Tk * ENCk + e;
  float a0 = 0.f, a1 = 0.f, a2 = 0.f, a3 = 0.f;                    // break dep chain
  for (int t = 0; t < Tk; t += 4) {
    a0 += s_aw[t + 0] * bf2f(eo[(size_t)(t + 0) * ENCk]);
    a1 += s_aw[t + 1] * bf2f(eo[(size_t)(t + 1) * ENCk]);
    a2 += s_aw[t + 2] * bf2f(eo[(size_t)(t + 2) * ENCk]);
    a3 += s_aw[t + 3] * bf2f(eo[(size_t)(t + 3) * ENCk]);
  }
  float acc = (a0 + a1) + (a2 + a3);
  ctx[b * ENCk + e] = acc;
  xbn[(size_t)b * 1536 + 1024 + e] = f2bf(acc);
}

// ---------------------------------------------------------------------------
extern "C" void kernel_launch(void* const* d_in, const int* in_sizes, int n_in,
                              void* d_out, int out_size, void* d_ws, size_t ws_size,
                              hipStream_t stream)
{
  const float* enc_out  = (const float*)d_in[0];
  const int*   enc_lens = (const int*)  d_in[1];
  const int*   ys       = (const int*)  d_in[2];
  const float* embed    = (const float*)d_in[3];
  const float* W_ih0    = (const float*)d_in[4];
  const float* W_hh0    = (const float*)d_in[5];
  const float* b0       = (const float*)d_in[6];
  const float* W_ih1    = (const float*)d_in[7];
  const float* W_hh1    = (const float*)d_in[8];
  const float* b1       = (const float*)d_in[9];
  const float* W_enc    = (const float*)d_in[10];
  const float* b_att    = (const float*)d_in[11];
  const float* v_att    = (const float*)d_in[12];
  const float* W_dec    = (const float*)d_in[13];
  const float* W_bn     = (const float*)d_in[14];
  const float* b_bn     = (const float*)d_in[15];
  const float* W_out    = (const float*)d_in[16];
  const float* b_out    = (const float*)d_in[17];
  float* out = (float*)d_out;
  (void)in_sizes; (void)n_in; (void)out_size; (void)ws_size;

  // ---- workspace bump allocator ----
  char* p = (char*)d_ws;
  auto alloc = [&](size_t bytes) -> void* {
    void* r = (void*)p; p += (bytes + 255) & ~(size_t)255; return r;
  };
  unsigned short* WT0   = (unsigned short*)alloc((size_t)4*Hk * 2048 * 2);  // 16 MB
  unsigned short* WT1   = (unsigned short*)alloc((size_t)4*Hk * 2048 * 2);  // 16 MB
  unsigned short* WdT   = (unsigned short*)alloc((size_t)Ak * Hk * 2);      // 1 MB
  unsigned short* WbnT  = (unsigned short*)alloc((size_t)Hk * 1536 * 2);    // 3 MB
  unsigned short* WoT   = (unsigned short*)alloc((size_t)Vk * Hk * 2);      // 20 MB
  unsigned short* WeT   = (unsigned short*)alloc((size_t)Ak * ENCk * 2);    // 0.5 MB
  unsigned short* eob   = (unsigned short*)alloc((size_t)Bk*Tk*ENCk * 2);   // 8 MB
  unsigned short* epb   = (unsigned short*)alloc((size_t)Bk*Tk*Ak * 2);     // 8 MB
  unsigned short* yemb  = (unsigned short*)alloc((size_t)STEPS*Bk*Ek * 2);  // 2 MB
  unsigned short* x0b   = (unsigned short*)alloc((size_t)Bk*2048 * 2);
  unsigned short* x1b   = (unsigned short*)alloc((size_t)Bk*2048 * 2);
  unsigned short* xbn   = (unsigned short*)alloc((size_t)Bk*1536 * 2);
  unsigned short* avb   = (unsigned short*)alloc((size_t)Bk*Hk * 2);
  float* g0   = (float*)alloc((size_t)Bk*4*Hk * 4);
  float* g1   = (float*)alloc((size_t)Bk*4*Hk * 4);
  float* c0   = (float*)alloc((size_t)Bk*Hk * 4);
  float* h0   = (float*)alloc((size_t)Bk*Hk * 4);
  float* c1   = (float*)alloc((size_t)Bk*Hk * 4);
  float* h1   = (float*)alloc((size_t)Bk*Hk * 4);
  float* ctx  = (float*)alloc((size_t)Bk*ENCk * 4);
  float* decW = (float*)alloc((size_t)Bk*Ak * 4);
  float* ebuf = (float*)alloc((size_t)Bk*Tk * 4);
  float* aw   = (float*)alloc((size_t)Bk*Tk * 4);

  // ---- one-time setup ----
  pack_lstm_k<<<(4*Hk*2048)/256, 256, 0, stream>>>(W_ih0, W_hh0, WT0);
  pack_lstm_k<<<(4*Hk*2048)/256, 256, 0, stream>>>(W_ih1, W_hh1, WT1);
  {
    long tot = (long)Ak * Hk;   // W_dec [1024][512] -> [512][1024]
    transpose_bf_k<<<(unsigned)((tot+255)/256), 256, 0, stream>>>(W_dec, WdT, Hk, Ak, tot);
  }
  {
    long tot = (long)Hk * 1536; // W_bn [1536][1024] -> [1024][1536]
    transpose_bf_k<<<(unsigned)((tot+255)/256), 256, 0, stream>>>(W_bn, WbnT, 1536, Hk, tot);
  }
  {
    long tot = (long)Vk * Hk;   // W_out [1024][10000] -> [10000][1024]
    transpose_bf_k<<<(unsigned)((tot+255)/256), 256, 0, stream>>>(W_out, WoT, Hk, Vk, tot);
  }
  {
    long tot = (long)Ak * ENCk; // W_enc [512][512] -> [512][512]
    transpose_bf_k<<<(unsigned)((tot+255)/256), 256, 0, stream>>>(W_enc, WeT, ENCk, Ak, tot);
  }
  {
    long tot = (long)Bk*Tk*ENCk;
    cvt_bf_k<<<(unsigned)((tot+255)/256), 256, 0, stream>>>(enc_out, eob, tot);
  }
  gather_emb_k<<<(STEPS*Bk*Ek + 255)/256, 256, 0, stream>>>(ys, embed, yemb);
  // enc_proj = enc_out @ W_enc + b_att  (M=8192, N=512, K=512) -> bf16
  wmma_gemm16_k<<<dim3(4, (Bk*Tk)/16), 256, 0, stream>>>(
      eob, ENCk, WeT, b_att, nullptr, epb, Ak, Ak, ENCk, 0);
  init_state_k<<<(Bk*Hk)/256, 256, 0, stream>>>(c0, h0, c1, h1, ctx, x1b);

  // ---- sequential decode: 129 steps ----
  for (int t = 0; t < STEPS; ++t) {
    build_x0_k<<<(Bk*2048)/256, 256, 0, stream>>>(yemb + (size_t)t*Bk*Ek, ctx, h0, x0b);
    // LSTM0 gates: [16,2048] x [2048,4096]
    wmma_gemm16_k<<<dim3(32,1), 256, 0, stream>>>(x0b, 2048, WT0, b0,
                                                  g0, nullptr, 4*Hk, 4*Hk, 2048, 0);
    lstm0_update_k<<<(Bk*Hk)/256, 256, 0, stream>>>(g0, c0, h0, x1b);
    // LSTM1 gates
    wmma_gemm16_k<<<dim3(32,1), 256, 0, stream>>>(x1b, 2048, WT1, b1,
                                                  g1, nullptr, 4*Hk, 4*Hk, 2048, 0);
    lstm1_update_k<<<(Bk*Hk)/256, 256, 0, stream>>>(g1, c1, h1, h0, x1b, xbn);
    // decW = dec @ W_dec : [16,1024] x [1024,512]
    wmma_gemm16_k<<<dim3(4,1), 256, 0, stream>>>(xbn, 1536, WdT, nullptr,
                                                 decW, nullptr, Ak, Ak, Hk, 0);
    att_e_k<<<(Bk*Tk)/8, 256, 0, stream>>>(epb, decW, v_att, ebuf);
    att_softmax_k<<<Bk, 256, 0, stream>>>(ebuf, enc_lens, aw);
    att_ctx_k<<<Bk, Tk, 0, stream>>>(aw, eob, ctx, xbn);
    // attn_v = tanh([dec,ctx] @ W_bn + b_bn) : [16,1536] x [1536,1024] -> bf16
    wmma_gemm16_k<<<dim3(8,1), 256, 0, stream>>>(xbn, 1536, WbnT, b_bn,
                                                 nullptr, avb, Hk, Hk, 1536, 1);
    // logits = attn_v @ W_out + b_out : [16,1024] x [1024,10000] -> d_out[:,t,:]
    wmma_gemm16_k<<<dim3(79,1), 256, 0, stream>>>(avb, Hk, WoT, b_out,
                                                  out + (size_t)t * Vk, nullptr,
                                                  STEPS * Vk, Vk, Hk, 0);
  }
}
